// BitConv2d_45964740001936
// MI455X (gfx1250) — matrix operand order repaired and checked by reading, not compiled, not run on hardware
//
#include <hip/hip_runtime.h>

typedef _Float16 h16;
typedef __attribute__((ext_vector_type(16))) _Float16 v16h;
typedef __attribute__((ext_vector_type(8)))  float    v8f;

#define BATCH 16
#define C_IN  128
#define C_OUT 128
#define HH    56
#define WW    56
#define HP    58
#define WP    58
#define KCH   36   // 9 taps * 128 ic / 32 K-per-WMMA
// xpad layout: [b][hp][wp][ic]  (f16, zero-padded borders)
// wswz layout: [chunk(36)][ocblock(8)][lane(32)][16 halves]  (WMMA A-fragment order)

#define XPAD_HALVES ((size_t)BATCH * HP * WP * C_IN)          // 6,889,472
#define XPAD_BYTES  (XPAD_HALVES * 2)                          // 13,778,944 (256B aligned)
#define WSWZ_HALVES ((size_t)KCH * 8 * 32 * 16)                // 147,456

// ---------------------------------------------------------------------------
// Prep 1: x (NCHW f32) -> xpad (b, hp, wp, ic) f16 with 1-pixel zero border.
// One WG per (b, hp). LDS transpose: coalesced global reads AND writes.
// ---------------------------------------------------------------------------
__global__ __launch_bounds__(256) void prep_x(const float* __restrict__ x,
                                              h16* __restrict__ xpad) {
    __shared__ h16 lds[C_IN * WW];                 // [ic][w], 14,336 B
    const int b   = blockIdx.x / HP;
    const int hp  = blockIdx.x % HP;
    const int tid = threadIdx.x;
    const bool inner = (hp >= 1) && (hp <= HH);
    if (inner) {
        const float* src = x + (size_t)b * C_IN * (HH * WW) + (size_t)(hp - 1) * WW;
        for (int e = tid; e < C_IN * WW; e += 256) {
            const int ic = e / WW, w = e % WW;     // consecutive lanes -> consecutive w
            lds[e] = (h16)src[(size_t)ic * (HH * WW) + w];
        }
    }
    __syncthreads();
    h16* dst = xpad + (size_t)((b * HP + hp) * WP) * C_IN;
    for (int e = tid; e < WP * C_IN; e += 256) {
        const int wp = e >> 7, ic = e & 127;       // consecutive lanes -> consecutive ic
        h16 v = (h16)0.f;
        if (inner && wp >= 1 && wp <= WW) v = lds[ic * WW + (wp - 1)];
        dst[e] = v;                                // fully coalesced f16 stores
    }
}

// ---------------------------------------------------------------------------
// Prep 2: W [oc][ic][kh][kw] f32 -> f16 A-fragments in WMMA 16x32 lane layout.
// A 16x32 f16 layout used here: lane<16 holds row M=lane, half j holds
// K = j (0..15); lane>=16 holds K = 16+j (16..31), contiguous like B.
// ---------------------------------------------------------------------------
__global__ __launch_bounds__(256) void prep_w(const float* __restrict__ Wt,
                                              h16* __restrict__ wswz) {
    const int i = blockIdx.x * 256 + threadIdx.x;
    if (i >= (int)WSWZ_HALVES) return;
    const int j    = i & 15;         // half index within lane's 32B fragment
    const int lane = (i >> 4) & 31;
    const int o    = (i >> 9) & 7;   // oc block
    const int c    = i >> 12;        // k-chunk
    const int klocal = j + ((lane & 16) ? 16 : 0);
    const int t  = c >> 2;                       // tap = kh*3+kw
    const int ic = (c & 3) * 32 + klocal;
    const int kh = t / 3, kw = t % 3;
    const int oc = o * 16 + (lane & 15);         // row M = lane%16
    wswz[i] = (h16)Wt[(((size_t)oc * C_IN + ic) * 3 + kh) * 3 + kw];
}

// ---------------------------------------------------------------------------
// Conv: one WG (8 wave32) per (batch, output row). Implicit GEMM:
//   D[oc, px] = sum_{tap, ic} Wf16[oc, tap*128+ic] * act[tap_dh][px+tap_dw][ic]
// Wave w owns oc block w; 4 accumulators cover the 56-pixel row (4x16, masked).
// Activation rows staged global->LDS with CDNA5 async-to-LDS (ASYNCcnt).
// ---------------------------------------------------------------------------
__global__ __launch_bounds__(256) void conv_wmma(const h16* __restrict__ xpad,
                                                 const h16* __restrict__ wswz,
                                                 const float* __restrict__ bias,
                                                 float* __restrict__ out) {
    __shared__ h16 act[3 * WP * C_IN];            // 44,544 B: 3 padded input rows
    const int b   = blockIdx.x / HH;
    const int h   = blockIdx.x % HH;
    const int tid = threadIdx.x;

    // Stage the 3 needed padded rows (one contiguous 44,544 B block of xpad)
    // using the direct global->LDS async datapath: no VGPR round-trip, one
    // drain wait on ASYNCcnt before the workgroup barrier.
    {
        const char* gsrc = (const char*)(xpad + (size_t)((b * HP + h) * WP) * C_IN);
        const unsigned lbase = (unsigned)(uintptr_t)(void*)&act[0];
        const int n128 = 3 * WP * C_IN * 2 / 16;  // 2784 x b128
        for (int i = tid; i < n128; i += 256) {
            const unsigned laddr = lbase + (unsigned)i * 16u;   // LDS byte address
            const char*    gaddr = gsrc + (size_t)i * 16;
            asm volatile("global_load_async_to_lds_b128 %0, %1, off"
                         :: "v"(laddr), "v"(gaddr)
                         : "memory");
        }
        asm volatile("s_wait_asynccnt 0x0" ::: "memory");
    }
    __syncthreads();

    const int wave  = tid >> 5;       // oc block
    const int lane  = tid & 31;
    const int n     = lane & 15;      // N column within tile
    const int khalf = lane & 16;      // B: lanes>=16 hold K=16..31 -> +16 ic

    v8f acc0 = {}, acc1 = {}, acc2 = {}, acc3 = {};
    const h16* wfrag = wswz + (size_t)(wave * 32 + lane) * 16;

    for (int c = 0; c < KCH; ++c) {
        v16h A = *(const v16h*)(wfrag + (size_t)c * (8 * 32 * 16));
        if (c + 1 < KCH)  // overlap L2->L0 latency of next A fragment with WMMAs
            __builtin_prefetch(wfrag + (size_t)(c + 1) * (8 * 32 * 16), 0, 3);
        const int t   = c >> 2;
        const int ic0 = (c & 3) * 32 + khalf;
        const int dh  = t / 3, dw = t % 3;
        const h16* brow = act + (size_t)(dh * WP) * C_IN + ic0;
        const int wc = n + dw;        // in-row column for tiles 0..2 (max 49 < 58)
        {   // tile 0: pixels 0..15
            v16h B = *(const v16h*)(brow + (size_t)wc * C_IN);
            acc0 = __builtin_amdgcn_wmma_f32_16x16x32_f16(false, A, false, B,
                                                          (short)0, acc0, false, false);
        }
        {   // tile 1: pixels 16..31
            v16h B = *(const v16h*)(brow + (size_t)(wc + 16) * C_IN);
            acc1 = __builtin_amdgcn_wmma_f32_16x16x32_f16(false, A, false, B,
                                                          (short)0, acc1, false, false);
        }
        {   // tile 2: pixels 32..47
            v16h B = *(const v16h*)(brow + (size_t)(wc + 32) * C_IN);
            acc2 = __builtin_amdgcn_wmma_f32_16x16x32_f16(false, A, false, B,
                                                          (short)0, acc2, false, false);
        }
        {   // tile 3: pixels 48..63 (48..55 valid; clamp reads, mask stores)
            int w3 = wc + 48; if (w3 > WP - 1) w3 = WP - 1;
            v16h B = *(const v16h*)(brow + (size_t)w3 * C_IN);
            acc3 = __builtin_amdgcn_wmma_f32_16x16x32_f16(false, A, false, B,
                                                          (short)0, acc3, false, false);
        }
    }

    // D layout (ISA 7.12.2): lane<16 -> N=lane, M=vgpr; lane>=16 -> M=vgpr+8.
    const int mbase = (lane & 16) ? 8 : 0;
    const int ocb   = wave * 16 + mbase;
    float bv[8];
#pragma unroll
    for (int v = 0; v < 8; ++v) bv[v] = bias[ocb + v];

#define STORE_TILE(ACC, TAU)                                                       \
    {                                                                              \
        const int wcol = (TAU)*16 + n;                                             \
        if (wcol < WW) {                                                           \
            _Pragma("unroll") for (int v = 0; v < 8; ++v) {                        \
                out[(((size_t)b * C_OUT + (ocb + v)) * HH + h) * WW + wcol] =      \
                    ACC[v] + bv[v];                                                \
            }                                                                      \
        }                                                                          \
    }
    STORE_TILE(acc0, 0)
    STORE_TILE(acc1, 1)
    STORE_TILE(acc2, 2)
    STORE_TILE(acc3, 3)
#undef STORE_TILE
}

extern "C" void kernel_launch(void* const* d_in, const int* in_sizes, int n_in,
                              void* d_out, int out_size, void* d_ws, size_t ws_size,
                              hipStream_t stream) {
    const float* x    = (const float*)d_in[0];
    const float* Wt   = (const float*)d_in[1];
    const float* bias = (const float*)d_in[2];
    float* out = (float*)d_out;

    h16* xpad = (h16*)d_ws;
    h16* wswz = (h16*)((char*)d_ws + XPAD_BYTES);   // 13,778,944 (256B aligned)

    prep_x<<<BATCH * HP, 256, 0, stream>>>(x, xpad);
    prep_w<<<(int)((WSWZ_HALVES + 255) / 256), 256, 0, stream>>>(Wt, wswz);
    conv_wmma<<<BATCH * HH, 256, 0, stream>>>(xpad, wswz, bias, out);
}
